// ConformerBlock_6966436954547
// MI455X (gfx1250) — matrix operand order repaired and checked by reading, not compiled, not run on hardware
//
#include <hip/hip_runtime.h>
#include <hip/hip_bf16.h>
#include <cstdint>
#include <cstddef>

// ---------------------------------------------------------------------------
// Conformer block for MI455X (gfx1250, wave32, WMMA + Tensor Data Mover).
// GEMMs: v_wmma_f32_16x16x32_bf16 with TDM (tensor_load_to_lds) double-buffered
// LDS staging (TENSORcnt-tracked async DMA with hardware row padding).
// Attention: flash-style with the relative-position bias computed as a WMMA
// GEMM into a 64x1025 LDS table (~173KB dynamic LDS; CDNA5 has 320KB/WGP).
// ---------------------------------------------------------------------------

typedef __attribute__((ext_vector_type(16))) __bf16 v16bf;
typedef __attribute__((ext_vector_type(8)))  __bf16 v8bf;
typedef __attribute__((ext_vector_type(8)))  float  v8f;
typedef __attribute__((ext_vector_type(4)))  unsigned uint4v;
typedef __attribute__((ext_vector_type(8)))  int     int8v;
typedef __attribute__((ext_vector_type(4)))  int     int4v;

#define DEV static __device__ __forceinline__

DEV float  bf2f(__bf16 v) { return (float)v; }
DEV __bf16 f2bf(float v)  { return (__bf16)v; }
DEV float  sigm(float x)  { return 1.0f / (1.0f + __expf(-x)); }

// Low 32 bits of a generic pointer into LDS = workgroup-relative LDS byte
// address (ISA 10.2 aperture mapping).
DEV unsigned lds_off(const void* p) { return (unsigned)(unsigned long long)p; }

// ---------------------------------------------------------------------------
// Tensor Data Mover: async 2D tile load global -> LDS (TENSORcnt).
// Builds the D# descriptor per CDNA5 ISA 8.3/8.4:
//   group0: [1:0]=count=1, [63:32]=lds_addr, [120:64]=global_addr, [127:126]=2
//   group1: data_size=2B, pad_enable, pad_interval/pad_amount,
//           tensor_dim0/1, tile_dim0/1, tensor_dim0_stride (elements)
// pad_interval: threshold 2^(i+1) DWORDs; pad_amount: (v+1) DWORDs.
// ---------------------------------------------------------------------------
DEV void tdm_load_2d(unsigned lds_addr, const void* gptr,
                     unsigned tile_d0, unsigned tile_d1,
                     unsigned long stride0_elems,
                     unsigned pad_interval, unsigned pad_amount) {
  const unsigned long long ga = (unsigned long long)gptr;
  uint4v g0;
  g0[0] = 1u;  // count=1, user-mode load descriptor
  g0[1] = lds_addr;
  g0[2] = (unsigned)(ga & 0xffffffffull);
  g0[3] = (unsigned)((ga >> 32) & 0x1ffffffull) | (2u << 30);  // type=2 (image)

  const unsigned td0 = tile_d0, td1 = tile_d1;  // tensor dims = tile dims (no OOB)
  int8v g1;
  g1[0] = (int)((1u << 16) | (1u << 20) |
                ((pad_interval & 7u) << 22) | ((pad_amount & 127u) << 25));
  g1[1] = (int)((td0 & 0xffffu) << 16);
  g1[2] = (int)(((td0 >> 16) & 0xffffu) | ((td1 & 0xffffu) << 16));
  g1[3] = (int)(((td1 >> 16) & 0xffffu) | ((tile_d0 & 0xffffu) << 16));
  g1[4] = (int)(tile_d1 & 0xffffu);  // tile_dim1; tile_dim2=0
  g1[5] = (int)(unsigned)(stride0_elems & 0xffffffffull);
  g1[6] = (int)(unsigned)((stride0_elems >> 32) & 0xffffull);
  g1[7] = 0;

  const int4v gz = {0, 0, 0, 0};
#if defined(__clang_major__) && (__clang_major__ >= 23)
  const int8v gz8 = {0, 0, 0, 0, 0, 0, 0, 0};
  __builtin_amdgcn_tensor_load_to_lds(g0, g1, gz, gz, gz8, 0);
#else
  __builtin_amdgcn_tensor_load_to_lds(g0, g1, gz, gz, 0);
#endif
}

// ---- CDNA5 WMMA fragment loaders (bf16, 16x16x32) -------------------------
// A matrix 16x32 (MxK): lanes 0-15 -> rows M=0..15 holding K {0..7,16..23};
// lanes 16-31 -> rows M=0..15 holding K {8..15,24..31}. (ISA 7.12.2)
DEV v16bf load_afrag(const __bf16* base, int ld, int row_base, int kbase, int lane) {
  const int r  = lane & 15;
  const int k0 = (lane >> 4) * 8;
  const __bf16* p = base + (long)(row_base + r) * ld + kbase + k0;
  v8bf lo = *(const v8bf*)(p);
  v8bf hi = *(const v8bf*)(p + 16);
  v16bf f;
#pragma unroll
  for (int i = 0; i < 8; ++i) { f[i] = lo[i]; f[8 + i] = hi[i]; }
  return f;
}

// B matrix 32x16 (KxN): lanes 0-15 -> cols N=0..15 holding K {0..7,16..23};
// lanes 16-31 -> cols N=0..15 holding K {8..15,24..31}.
DEV v16bf load_bfrag(const __bf16* base, int ld, int krow_base, int col_base, int lane) {
  const int n  = lane & 15;
  const int k0 = (lane >> 4) * 8;
  const __bf16* p = base + (long)(krow_base + k0) * ld + col_base + n;
  v16bf f;
#pragma unroll
  for (int i = 0; i < 8; ++i) { f[i] = p[(long)i * ld]; f[8 + i] = p[(long)(16 + i) * ld]; }
  return f;
}

DEV v8f wmma_bf16(v16bf a, v16bf b, v8f c) {
  return __builtin_amdgcn_wmma_f32_16x16x32_bf16(false, a, false, b, (short)0, c, false, false);
}

// ---------------------------------------------------------------------------
// f32 -> bf16 conversion
// ---------------------------------------------------------------------------
__global__ void k_cvt(const float* __restrict__ s, __bf16* __restrict__ d, long n) {
  long i = (long)blockIdx.x * blockDim.x + threadIdx.x;
  if (i < n) d[i] = f2bf(s[i]);
}

// ---------------------------------------------------------------------------
// LayerNorm over D=512, one wave32 per row, 8 rows per block.
// ---------------------------------------------------------------------------
__global__ __launch_bounds__(256)
void k_layernorm(const float* __restrict__ x, const float* __restrict__ g,
                 const float* __restrict__ bta, __bf16* __restrict__ outb,
                 float* __restrict__ outf, int rows) {
  const int w = threadIdx.x >> 5, lane = threadIdx.x & 31;
  const int row = blockIdx.x * 8 + w;
  if (row >= rows) return;
  const float* xr = x + (long)row * 512;
  float v[16];
  float s = 0.0f;
#pragma unroll
  for (int i = 0; i < 16; ++i) { v[i] = xr[lane + i * 32]; s += v[i]; }
#pragma unroll
  for (int m = 16; m >= 1; m >>= 1) s += __shfl_xor(s, m, 32);
  const float mean = s * (1.0f / 512.0f);
  float vs = 0.0f;
#pragma unroll
  for (int i = 0; i < 16; ++i) { float d0 = v[i] - mean; vs += d0 * d0; }
#pragma unroll
  for (int m = 16; m >= 1; m >>= 1) vs += __shfl_xor(vs, m, 32);
  const float rstd = rsqrtf(vs * (1.0f / 512.0f) + 1e-5f);
#pragma unroll
  for (int i = 0; i < 16; ++i) {
    const int c = lane + i * 32;
    const float y = (v[i] - mean) * rstd * g[c] + bta[c];
    outb[(long)row * 512 + c] = f2bf(y);
    if (outf) outf[(long)row * 512 + c] = y;
  }
}

// ---------------------------------------------------------------------------
// Tiled WMMA GEMM with TDM double-buffered LDS staging.
// C[M,Nc] = A[M,Kd] @ B[Kd,Nc] (+bias), fused epilogue.
// 128x128x32 tile, 256 threads (8 waves, 2x4 grid; 64x32 per wave).
// Wave 0 issues tensor_load_to_lds for stage s+1 while all waves compute
// stage s; s_wait_tensorcnt(0) + barrier publishes the buffer.
// MODE: 0 f32, 1 bf16, 2 swish->bf16, 3 res+acc->f32, 4 res+0.5*acc->f32
// ---------------------------------------------------------------------------
template <int MODE>
__global__ __launch_bounds__(256)
void k_gemm(const __bf16* __restrict__ A, const __bf16* __restrict__ Bm,
            const float* __restrict__ bias, const float* __restrict__ res,
            float* __restrict__ outf, __bf16* __restrict__ outb,
            int M, int Kd, int Nc) {
  __shared__ __bf16 As[2 * 128 * 40];   // [buf][row][k],  row stride 40 (pad +8)
  __shared__ __bf16 Bs[2 * 32 * 136];   // [buf][k][n],    row stride 136 (pad +8)
  const int tid = threadIdx.x, lane = tid & 31;
  const int w = tid >> 5;
  const int wm = (w >> 2) * 64;
  const int wn = (w & 3) * 32;
  const int m0 = blockIdx.y * 128, n0 = blockIdx.x * 128;
  const int half = lane >> 4, ncol = lane & 15;

  v8f acc[4][2];
  const v8f vz = {0, 0, 0, 0, 0, 0, 0, 0};
#pragma unroll
  for (int i = 0; i < 4; ++i)
#pragma unroll
    for (int j = 0; j < 2; ++j) acc[i][j] = vz;

  const int nstages = Kd >> 5;

  // issue TDM loads for stage st into buffer buf (wave 0 only)
  auto issue = [&](int st, int buf) {
    if (w == 0) {
      // A tile: 128 rows x 32 elems, row = 16 DW -> pad_interval 3, +4 DW pad
      tdm_load_2d(lds_off(As + buf * 128 * 40),
                  A + (long)m0 * Kd + (long)st * 32,
                  /*tile_d0=*/32, /*tile_d1=*/128, /*stride0=*/(unsigned long)Kd,
                  /*pad_interval=*/3, /*pad_amount=*/3);
      // B tile: 32 rows x 128 elems, row = 64 DW -> pad_interval 5, +4 DW pad
      tdm_load_2d(lds_off(Bs + buf * 32 * 136),
                  Bm + (long)st * 32 * Nc + n0,
                  /*tile_d0=*/128, /*tile_d1=*/32, /*stride0=*/(unsigned long)Nc,
                  /*pad_interval=*/5, /*pad_amount=*/3);
    }
  };

  issue(0, 0);
  __builtin_amdgcn_s_wait_tensorcnt(0);
  __syncthreads();

  for (int st = 0; st < nstages; ++st) {
    const int cur = st & 1;
    if (st + 1 < nstages) issue(st + 1, cur ^ 1);

    const __bf16* Ab = As + cur * 128 * 40;
    const __bf16* Bb = Bs + cur * 32 * 136;
    v16bf bf[2];
#pragma unroll
    for (int tn = 0; tn < 2; ++tn) bf[tn] = load_bfrag(Bb, 136, 0, wn + tn * 16, lane);
#pragma unroll
    for (int tm = 0; tm < 4; ++tm) {
      v16bf af = load_afrag(Ab, 40, wm + tm * 16, 0, lane);
#pragma unroll
      for (int tn = 0; tn < 2; ++tn) acc[tm][tn] = wmma_bf16(af, bf[tn], acc[tm][tn]);
    }

    __builtin_amdgcn_s_wait_tensorcnt(0);  // wave0: stage st+1 DMA complete
    __syncthreads();                       // publish buffer / retire reads
  }

  // Epilogue: C layout lane l, vgpr g -> row = g + 8*(l>>4), col = l&15.
#pragma unroll
  for (int tm = 0; tm < 4; ++tm) {
#pragma unroll
    for (int tn = 0; tn < 2; ++tn) {
      const int gcol = n0 + wn + tn * 16 + ncol;
      const float bv = bias ? bias[gcol] : 0.0f;
#pragma unroll
      for (int g = 0; g < 8; ++g) {
        const int grow = m0 + wm + tm * 16 + g + 8 * half;
        const long idx = (long)grow * Nc + gcol;
        const float v = acc[tm][tn][g] + bv;
        if (MODE == 0)      outf[idx] = v;
        else if (MODE == 1) outb[idx] = f2bf(v);
        else if (MODE == 2) outb[idx] = f2bf(v * sigm(v));
        else if (MODE == 3) outf[idx] = res[idx] + v;
        else if (MODE == 4) outf[idx] = res[idx] + 0.5f * v;
      }
    }
  }
}

// ---------------------------------------------------------------------------
// GLU: h[r, :1024] * sigmoid(h[r, 1024:])   (h is [8192, 2048] bf16)
// ---------------------------------------------------------------------------
__global__ void k_glu(const __bf16* __restrict__ h, __bf16* __restrict__ o, long total) {
  long i = (long)blockIdx.x * blockDim.x + threadIdx.x;
  if (i >= total) return;
  const long r = i >> 10;
  const int c  = (int)(i & 1023);
  const float a  = bf2f(h[r * 2048 + c]);
  const float gt = bf2f(h[r * 2048 + 1024 + c]);
  o[i] = f2bf(a * sigm(gt));
}

// ---------------------------------------------------------------------------
// Depthwise conv (K=31, zero pad 15/15) + BN(eval) + swish. [B,N,CI] layout.
// ---------------------------------------------------------------------------
__global__ void k_dwconv(const __bf16* __restrict__ g, const float* __restrict__ dw,
                         const float* __restrict__ db, const float* __restrict__ bn_g,
                         const float* __restrict__ bn_b, const float* __restrict__ bn_m,
                         const float* __restrict__ bn_v, __bf16* __restrict__ o) {
  const long i = (long)blockIdx.x * blockDim.x + threadIdx.x;  // B*N*CI threads
  const int c = (int)(i & 1023);
  const int n = (int)((i >> 10) & 1023);
  const int b = (int)(i >> 20);
  float acc = 0.0f;
#pragma unroll
  for (int k = 0; k < 31; ++k) {
    const int nn = n + k - 15;
    if (nn >= 0 && nn < 1024)
      acc += bf2f(g[(((long)b << 10) + nn) * 1024 + c]) * dw[c * 31 + k];
  }
  acc += db[c];
  acc = (acc - bn_m[c]) * rsqrtf(bn_v[c] + 1e-5f) * bn_g[c] + bn_b[c];
  o[i] = f2bf(acc * sigm(acc));
}

// ---------------------------------------------------------------------------
// Flash attention with relative positional bias.
// One workgroup = (b, h, 64-row block). 128 threads = 4 wave32 waves,
// wave w owns rows [16w, 16w+16).
// Phase 1: qr = Q_blk @ rel^T (64 x 1025) via WMMA -> LDS (bf16).
// Phase 2: streaming softmax over 16 column blocks of 64;
//          S = Q K^T (WMMA), bias gathered from qr by dist, O += P V (WMMA).
// ---------------------------------------------------------------------------
__global__ __launch_bounds__(128)
void k_attn(const __bf16* __restrict__ qbf, const __bf16* __restrict__ kvbf,
            const __bf16* __restrict__ relbf, __bf16* __restrict__ obf) {
  constexpr int N = 1024, HD = 512, KVW = 1024, QRLD = 1096;
  const int n0 = blockIdx.x * 64;
  const int h  = blockIdx.y;
  const int b  = blockIdx.z;
  const int tid = threadIdx.x, lane = tid & 31, w = tid >> 5;
  const int half = lane >> 4, ncol = lane & 15;
  const int wr = w * 16;

  extern __shared__ char smem_raw[];
  __bf16* Qs = (__bf16*)smem_raw;     // [64][72]   (r, d)
  __bf16* Kt = Qs + 64 * 72;          // [64][72]   transposed (d, c) / rel (d, j)
  __bf16* Vs = Kt + 64 * 72;          // [64][72]   (c, d)
  __bf16* Ps = Vs + 64 * 72;          // [64][72]   (r, c)
  __bf16* qr = Ps + 64 * 72;          // [64][1096] (r, j)

  const long qrow0 = (long)b * N + n0;

  for (int u = tid; u < 64 * 8; u += 128) {
    const int r = u >> 3, s = u & 7;
    v8bf v = *(const v8bf*)(qbf + (qrow0 + r) * HD + h * 64 + s * 8);
    *(v8bf*)(Qs + r * 72 + s * 8) = v;
  }
  __syncthreads();

  // ---- phase 1: qr[r][j] = sum_d Q[r][d] * rel[j][d] ----
  for (int jb = 0; jb < 17; ++jb) {
    __syncthreads();
    for (int u = tid; u < 64 * 64; u += 128) {
      const int jloc = u >> 6, d = u & 63;
      int jg = jb * 64 + jloc;
      if (jg > 1024) jg = 1024;
      Kt[d * 72 + jloc] = relbf[jg * 64 + d];
    }
    __syncthreads();
#pragma unroll
    for (int jj = 0; jj < 4; ++jj) {
      v8f acc = {0, 0, 0, 0, 0, 0, 0, 0};
#pragma unroll
      for (int kk = 0; kk < 64; kk += 32) {
        v16bf a  = load_afrag(Qs, 72, wr, kk, lane);
        v16bf bb = load_bfrag(Kt, 72, kk, jj * 16, lane);
        acc = wmma_bf16(a, bb, acc);
      }
#pragma unroll
      for (int g = 0; g < 8; ++g) {
        const int row = wr + g + 8 * half;
        const int col = jb * 64 + jj * 16 + ncol;
        qr[row * QRLD + col] = f2bf(acc[g]);
      }
    }
  }

  // ---- phase 2: streaming softmax ----
  float mi[8], li[8];
  v8f Oa[4];
  const v8f vz = {0, 0, 0, 0, 0, 0, 0, 0};
#pragma unroll
  for (int g = 0; g < 8; ++g) { mi[g] = -1e30f; li[g] = 0.0f; }
#pragma unroll
  for (int dd = 0; dd < 4; ++dd) Oa[dd] = vz;

  for (int cb = 0; cb < 16; ++cb) {
    __syncthreads();
    for (int u = tid; u < 64 * 64; u += 128) {
      const int c = u >> 6, d = u & 63;
      Kt[d * 72 + c] = kvbf[((long)b * N + cb * 64 + c) * KVW + h * 64 + d];
    }
    for (int u = tid; u < 64 * 8; u += 128) {
      const int c = u >> 3, s = u & 7;
      v8bf v = *(const v8bf*)(kvbf + ((long)b * N + cb * 64 + c) * KVW + 512 + h * 64 + s * 8);
      *(v8bf*)(Vs + c * 72 + s * 8) = v;
    }
    __syncthreads();

    v8f sa[4];
#pragma unroll
    for (int nn = 0; nn < 4; ++nn) {
      sa[nn] = vz;
#pragma unroll
      for (int kk = 0; kk < 64; kk += 32) {
        v16bf a  = load_afrag(Qs, 72, wr, kk, lane);
        v16bf bb = load_bfrag(Kt, 72, kk, nn * 16, lane);
        sa[nn] = wmma_bf16(a, bb, sa[nn]);
      }
    }

    float sv[4][8];
#pragma unroll
    for (int nn = 0; nn < 4; ++nn)
#pragma unroll
      for (int g = 0; g < 8; ++g) {
        const int rloc = wr + g + 8 * half;
        const int n_abs = n0 + rloc;
        const int c_abs = cb * 64 + nn * 16 + ncol;
        int dist = n_abs - c_abs;
        dist = dist < -512 ? -512 : (dist > 512 ? 512 : dist);
        dist += 512;
        const float bias = bf2f(qr[rloc * QRLD + dist]);
        sv[nn][g] = (sa[nn][g] + bias) * 0.125f;  // DH^-0.5
      }

    float rmax[8];
#pragma unroll
    for (int g = 0; g < 8; ++g) {
      float m = sv[0][g];
      m = fmaxf(m, sv[1][g]); m = fmaxf(m, sv[2][g]); m = fmaxf(m, sv[3][g]);
#pragma unroll
      for (int msk = 1; msk <= 8; msk <<= 1) m = fmaxf(m, __shfl_xor(m, msk, 32));
      rmax[g] = m;
    }

    float mnew[8], fac[8], rsum[8];
#pragma unroll
    for (int g = 0; g < 8; ++g) {
      mnew[g] = fmaxf(mi[g], rmax[g]);
      fac[g]  = __expf(mi[g] - mnew[g]);
      rsum[g] = 0.0f;
    }

    float p[4][8];
#pragma unroll
    for (int nn = 0; nn < 4; ++nn)
#pragma unroll
      for (int g = 0; g < 8; ++g) {
        p[nn][g] = __expf(sv[nn][g] - mnew[g]);
        rsum[g] += p[nn][g];
      }
#pragma unroll
    for (int g = 0; g < 8; ++g) {
#pragma unroll
      for (int msk = 1; msk <= 8; msk <<= 1) rsum[g] += __shfl_xor(rsum[g], msk, 32);
      li[g] = li[g] * fac[g] + rsum[g];
      mi[g] = mnew[g];
    }
#pragma unroll
    for (int dd = 0; dd < 4; ++dd)
#pragma unroll
      for (int g = 0; g < 8; ++g) Oa[dd][g] *= fac[g];

#pragma unroll
    for (int nn = 0; nn < 4; ++nn)
#pragma unroll
      for (int g = 0; g < 8; ++g)
        Ps[(wr + g + 8 * half) * 72 + nn * 16 + ncol] = f2bf(p[nn][g]);

#pragma unroll
    for (int dd = 0; dd < 4; ++dd)
#pragma unroll
      for (int kk = 0; kk < 64; kk += 32) {
        v16bf a  = load_afrag(Ps, 72, wr, kk, lane);
        v16bf bb = load_bfrag(Vs, 72, kk, dd * 16, lane);
        Oa[dd] = wmma_bf16(a, bb, Oa[dd]);
      }
  }

#pragma unroll
  for (int g = 0; g < 8; ++g) {
    const float inv = 1.0f / li[g];
    const long row = qrow0 + wr + g + 8 * half;
#pragma unroll
    for (int dd = 0; dd < 4; ++dd)
      obf[row * HD + h * 64 + dd * 16 + ncol] = f2bf(Oa[dd][g] * inv);
  }
}

// ---------------------------------------------------------------------------
// Host orchestration
// ---------------------------------------------------------------------------
extern "C" void kernel_launch(void* const* d_in, const int* in_sizes, int n_in,
                              void* d_out, int out_size, void* d_ws, size_t ws_size,
                              hipStream_t stream) {
  const float* x     = (const float*)d_in[0];
  const float* f1_g  = (const float*)d_in[1];
  const float* f1_b  = (const float*)d_in[2];
  const float* f1_w1 = (const float*)d_in[3];
  const float* f1_b1 = (const float*)d_in[4];
  const float* f1_w2 = (const float*)d_in[5];
  const float* f1_b2 = (const float*)d_in[6];
  const float* a_g   = (const float*)d_in[7];
  const float* a_b   = (const float*)d_in[8];
  const float* wq    = (const float*)d_in[9];
  const float* wkv   = (const float*)d_in[10];
  const float* wo    = (const float*)d_in[11];
  const float* wo_b  = (const float*)d_in[12];
  const float* rel   = (const float*)d_in[13];
  const float* c_g   = (const float*)d_in[14];
  const float* c_b   = (const float*)d_in[15];
  const float* cw1   = (const float*)d_in[16];
  const float* cb1   = (const float*)d_in[17];
  const float* dwp   = (const float*)d_in[18];
  const float* dbp   = (const float*)d_in[19];
  const float* bn_g  = (const float*)d_in[20];
  const float* bn_b  = (const float*)d_in[21];
  const float* bn_m  = (const float*)d_in[22];
  const float* bn_v  = (const float*)d_in[23];
  const float* cw2   = (const float*)d_in[24];
  const float* cb2   = (const float*)d_in[25];
  const float* f2_g  = (const float*)d_in[26];
  const float* f2_b  = (const float*)d_in[27];
  const float* f2_w1 = (const float*)d_in[28];
  const float* f2_b1 = (const float*)d_in[29];
  const float* f2_w2 = (const float*)d_in[30];
  const float* f2_b2 = (const float*)d_in[31];
  const float* p_g   = (const float*)d_in[32];
  const float* p_b   = (const float*)d_in[33];
  (void)in_sizes; (void)n_in; (void)out_size; (void)ws_size;

  const long R = 8192;  // B*N
  char* ws = (char*)d_ws;
  auto alloc = [&](size_t bytes) -> char* {
    char* p = ws;
    ws += (bytes + 255) & ~(size_t)255;
    return p;
  };
  float*  xcur  = (float*)alloc(R * 512 * 4);
  __bf16* hbf   = (__bf16*)alloc(R * 512 * 2);
  __bf16* bigbf = (__bf16*)alloc(R * 2048 * 2);   // FF acts / cw1 out / attn O
  __bf16* glubf = (__bf16*)alloc(R * 1024 * 2);
  __bf16* qb    = (__bf16*)alloc(R * 512 * 2);
  __bf16* kvb   = (__bf16*)alloc(R * 1024 * 2);   // later reused as conv out
  __bf16* wb_f1w1 = (__bf16*)alloc(512 * 2048 * 2);
  __bf16* wb_f1w2 = (__bf16*)alloc(2048 * 512 * 2);
  __bf16* wb_wq   = (__bf16*)alloc(512 * 512 * 2);
  __bf16* wb_wkv  = (__bf16*)alloc(512 * 1024 * 2);
  __bf16* wb_wo   = (__bf16*)alloc(512 * 512 * 2);
  __bf16* wb_cw1  = (__bf16*)alloc(512 * 2048 * 2);
  __bf16* wb_cw2  = (__bf16*)alloc(1024 * 512 * 2);
  __bf16* wb_f2w1 = (__bf16*)alloc(512 * 2048 * 2);
  __bf16* wb_f2w2 = (__bf16*)alloc(2048 * 512 * 2);
  __bf16* relb    = (__bf16*)alloc(1025 * 64 * 2);

  auto cvt = [&](const float* s, __bf16* d, long n) {
    k_cvt<<<(unsigned)((n + 255) / 256), 256, 0, stream>>>(s, d, n);
  };
  cvt(f1_w1, wb_f1w1, 512L * 2048);
  cvt(f1_w2, wb_f1w2, 2048L * 512);
  cvt(wq,    wb_wq,   512L * 512);
  cvt(wkv,   wb_wkv,  512L * 1024);
  cvt(wo,    wb_wo,   512L * 512);
  cvt(cw1,   wb_cw1,  512L * 2048);
  cvt(cw2,   wb_cw2,  1024L * 512);
  cvt(f2_w1, wb_f2w1, 512L * 2048);
  cvt(f2_w2, wb_f2w2, 2048L * 512);
  cvt(rel,   relb,    1025L * 64);

  // ---- FF1 (half-step) ----
  k_layernorm<<<1024, 256, 0, stream>>>(x, f1_g, f1_b, hbf, nullptr, 8192);
  k_gemm<2><<<dim3(16, 64), 256, 0, stream>>>(hbf, wb_f1w1, f1_b1, nullptr,
                                              nullptr, bigbf, 8192, 512, 2048);
  k_gemm<4><<<dim3(4, 64), 256, 0, stream>>>(bigbf, wb_f1w2, f1_b2, x,
                                             xcur, nullptr, 8192, 2048, 512);

  // ---- Attention ----
  k_layernorm<<<1024, 256, 0, stream>>>(xcur, a_g, a_b, hbf, nullptr, 8192);
  k_gemm<1><<<dim3(4, 64), 256, 0, stream>>>(hbf, wb_wq, nullptr, nullptr,
                                             nullptr, qb, 8192, 512, 512);
  k_gemm<1><<<dim3(8, 64), 256, 0, stream>>>(hbf, wb_wkv, nullptr, nullptr,
                                             nullptr, kvb, 8192, 512, 1024);
  const size_t attn_smem = (size_t)(4 * 64 * 72 + 64 * 1096) * 2;  // ~173KB
  k_attn<<<dim3(16, 8, 8), 128, attn_smem, stream>>>(qb, kvb, relb, bigbf);
  k_gemm<3><<<dim3(4, 64), 256, 0, stream>>>(bigbf, wb_wo, wo_b, xcur,
                                             xcur, nullptr, 8192, 512, 512);

  // ---- Conv module ----
  k_layernorm<<<1024, 256, 0, stream>>>(xcur, c_g, c_b, hbf, nullptr, 8192);
  k_gemm<1><<<dim3(16, 64), 256, 0, stream>>>(hbf, wb_cw1, cb1, nullptr,
                                              nullptr, bigbf, 8192, 512, 2048);
  k_glu<<<(unsigned)((R * 1024 + 255) / 256), 256, 0, stream>>>(bigbf, glubf, R * 1024);
  k_dwconv<<<(unsigned)((R * 1024) / 256), 256, 0, stream>>>(
      glubf, dwp, dbp, bn_g, bn_b, bn_m, bn_v, kvb);
  k_gemm<3><<<dim3(4, 64), 256, 0, stream>>>(kvb, wb_cw2, cb2, xcur,
                                             xcur, nullptr, 8192, 1024, 512);

  // ---- FF2 (half-step) + post-LN ----
  k_layernorm<<<1024, 256, 0, stream>>>(xcur, f2_g, f2_b, hbf, nullptr, 8192);
  k_gemm<2><<<dim3(16, 64), 256, 0, stream>>>(hbf, wb_f2w1, f2_b1, nullptr,
                                              nullptr, bigbf, 8192, 512, 2048);
  k_gemm<4><<<dim3(4, 64), 256, 0, stream>>>(bigbf, wb_f2w2, f2_b2, xcur,
                                             xcur, nullptr, 8192, 2048, 512);
  k_layernorm<<<1024, 256, 0, stream>>>(xcur, p_g, p_b, hbf, (float*)d_out, 8192);
}
